// Performer_88295937671722
// MI455X (gfx1250) — matrix operand order repaired and checked by reading, not compiled, not run on hardware
//
#include <hip/hip_runtime.h>
#include <math.h>

// ---------------------------------------------------------------------------
// Performer causal linear attention, chunked formulation for gfx1250 (wave32,
// WMMA + Tensor Data Mover).  B=4, S=2048, D=128, chunk=128, 16 chunks/batch.
// ---------------------------------------------------------------------------
static constexpr int DIM   = 128;   // feature / head dim
static constexpr int CH    = 128;   // chunk length (tokens)
static constexpr int NCK   = 16;    // chunks per batch (S / CH)
static constexpr int NB    = 4;     // batch
static constexpr int NS    = 2048;  // sequence length

typedef __attribute__((ext_vector_type(16))) __bf16 v16bf;
typedef __attribute__((ext_vector_type(8)))  float  v8f;
typedef __attribute__((ext_vector_type(4)))  unsigned int v4u;
typedef __attribute__((ext_vector_type(8)))  int v8i;
typedef __attribute__((ext_vector_type(4)))  int v4i;

union FragU { v16bf v; unsigned short u[16]; };

__device__ __forceinline__ unsigned short f2bf(float x) {
  unsigned int b = __float_as_uint(x);
  b += 0x7FFFu + ((b >> 16) & 1u);          // round-to-nearest-even
  return (unsigned short)(b >> 16);
}
__device__ __forceinline__ float bf2f(unsigned short h) {
  return __uint_as_float(((unsigned int)h) << 16);
}
__device__ __forceinline__ v8f vzero8() {
  v8f z = {0.f, 0.f, 0.f, 0.f, 0.f, 0.f, 0.f, 0.f};
  return z;
}
__device__ __forceinline__ v8f wmma_bf16(v16bf a, v16bf b, v8f c) {
  // D(f32 16x16) = A(bf16 16x32) * B(bf16 32x16) + C
  return __builtin_amdgcn_wmma_f32_16x16x32_bf16(false, a, false, b,
                                                 (short)0, c, false, false);
}

// --- Tensor Data Mover: contiguous 32 KB (16384 x bf16) tile -> LDS --------
// 1-D descriptor: data_size=2B, tile_dim0=tensor_dim0=stride=16384, groups
// 2/3 zero (tile_dim1/2/3/4 = 0 -> unused).  Issue from ONE wave only.
__device__ __forceinline__ unsigned lds_addr_of(const void* p) {
  return (unsigned)(unsigned long long)p;   // LDS aperture: addr[31:0] = offset
}
__device__ __forceinline__ void tdm_load_chunk(const void* gsrc,
                                               unsigned lds_off) {
  const unsigned long long ga = (unsigned long long)gsrc;
  v4u g0;
  g0[0] = 1u;                                  // count=1 (valid user D#)
  g0[1] = lds_off;                             // lds_addr (bytes)
  g0[2] = (unsigned)ga;                        // global_addr[31:0]
  g0[3] = (unsigned)((ga >> 32) & 0x01FFFFFFu) // global_addr[56:32]
          | (2u << 30);                        // type = 2 ("image")
  v8i g1;
  g1[0] = (int)(1u << 16);       // wg_mask=0, data_size=1 (2 bytes)
  g1[1] = (int)(16384u << 16);   // tensor_dim0[15:0]=16384 (hi half of dword)
  g1[2] = (int)(1u << 16);       // tensor_dim0[31:16]=0 ; tensor_dim1[15:0]=1
  g1[3] = (int)(16384u << 16);   // tensor_dim1[31:16]=0 ; tile_dim0=16384
  g1[4] = 0;                     // tile_dim1=0 (unused), tile_dim2=0
  g1[5] = 16384;                 // tensor_dim0_stride[31:0]
  g1[6] = 0;                     // stride hi, tensor_dim1_stride lo
  g1[7] = 0;
  const v4i gz = {0, 0, 0, 0};
#if defined(__clang_major__) && __clang_major__ >= 23
  const v8i gz8 = {0, 0, 0, 0, 0, 0, 0, 0};
  __builtin_amdgcn_tensor_load_to_lds(g0, g1, gz, gz, gz8, 0);
#else
  __builtin_amdgcn_tensor_load_to_lds(g0, g1, gz, gz, 0);
#endif
}

// --- fragment gathers from LDS, per CDNA5 ISA 7.12.2 VGPR layouts ----------
// A (16x32, row-major  A[m][k] = lds[(m0+m)*ld + k0+k])
__device__ __forceinline__ v16bf frag_A(const unsigned short* lds, int m0,
                                        int k0, int ld, int lane) {
  FragU f;
  const int m = lane & 15, half = lane >> 4;
  const unsigned short* base = lds + (m0 + m) * ld + k0;
#pragma unroll
  for (int i = 0; i < 8; ++i) {
    const int k = ((i >> 2) << 4) + (half << 3) + ((i & 3) << 1);
    f.u[2 * i]     = base[k];
    f.u[2 * i + 1] = base[k + 1];
  }
  return f.v;
}
// A-transposed (A[m][k] = lds[(k0+k)*ld + m0+m])
__device__ __forceinline__ v16bf frag_AT(const unsigned short* lds, int m0,
                                         int k0, int ld, int lane) {
  FragU f;
  const int m = lane & 15, half = lane >> 4;
#pragma unroll
  for (int i = 0; i < 8; ++i) {
    const int k = ((i >> 2) << 4) + (half << 3) + ((i & 3) << 1);
    f.u[2 * i]     = lds[(k0 + k) * ld + m0 + m];
    f.u[2 * i + 1] = lds[(k0 + k + 1) * ld + m0 + m];
  }
  return f.v;
}
// B (32x16, row-major  B[k][n] = lds[(k0+k)*ld + n0+n])
__device__ __forceinline__ v16bf frag_B(const unsigned short* lds, int k0,
                                        int n0, int ld, int lane) {
  FragU f;
  const int n = lane & 15, half = lane >> 4;
#pragma unroll
  for (int i = 0; i < 8; ++i) {
    const int k = ((i >> 2) << 4) + (half << 3) + ((i & 3) << 1);
    f.u[2 * i]     = lds[(k0 + k) * ld + n0 + n];
    f.u[2 * i + 1] = lds[(k0 + k + 1) * ld + n0 + n];
  }
  return f.v;
}
// B-transposed (B[k][n] = lds[(n0+n)*ld + k0+k])
__device__ __forceinline__ v16bf frag_BT(const unsigned short* lds, int n0,
                                         int k0, int ld, int lane) {
  FragU f;
  const int n = lane & 15, half = lane >> 4;
  const unsigned short* base = lds + (n0 + n) * ld + k0;
#pragma unroll
  for (int i = 0; i < 8; ++i) {
    const int k = ((i >> 2) << 4) + (half << 3) + ((i & 3) << 1);
    f.u[2 * i]     = base[k];
    f.u[2 * i + 1] = base[k + 1];
  }
  return f.v;
}

// ---------------------------------------------------------------------------
// Kernel: f32 -> bf16 cast (V)
// ---------------------------------------------------------------------------
__global__ void __launch_bounds__(256) perf_cast_bf16(
    const float* __restrict__ x, unsigned short* __restrict__ y, int n) {
  const int i = blockIdx.x * 256 + (int)threadIdx.x;
  if (i < n) y[i] = f2bf(x[i]);
}

// ---------------------------------------------------------------------------
// Kernel: feature map  phi = exp(X @ W^T - 0.5*||x||) / sqrt(D), bf16 out.
// 64 rows per block; proj via WMMA (A = X rows, B[k][n] = W[n][k]).
// (Vector loads kept here: the f32->bf16 conversion and row norms must be
//  computed in-flight, which TDM cannot do.)
// ---------------------------------------------------------------------------
__global__ void __launch_bounds__(256) perf_featurize(
    const float* __restrict__ X, const float* __restrict__ W,
    unsigned short* __restrict__ phiOut) {
  __shared__ unsigned short sX[64 * DIM];     // 16 KB
  __shared__ unsigned short sW[DIM * DIM];    // 32 KB
  __shared__ float spart[256];
  __shared__ float snorm[64];
  const int tid = (int)threadIdx.x;
  const long rowbase = (long)blockIdx.x * 64;
  {  // load X rows (f32), convert to bf16, accumulate row sum-of-squares
    const int r = tid >> 2, q = tid & 3;
    const float* src = X + (rowbase + r) * DIM + q * 32;
    unsigned short* dst = sX + r * DIM + q * 32;
    float ss = 0.f;
#pragma unroll
    for (int i = 0; i < 32; ++i) {
      const float v = src[i];
      ss += v * v;
      dst[i] = f2bf(v);
    }
    spart[tid] = ss;
  }
  for (int idx = tid; idx < DIM * DIM; idx += 256) sW[idx] = f2bf(W[idx]);
  __syncthreads();
  if (tid < 64)
    snorm[tid] = sqrtf(spart[4 * tid] + spart[4 * tid + 1] +
                       spart[4 * tid + 2] + spart[4 * tid + 3]);
  __syncthreads();

  const int wave = tid >> 5, lane = tid & 31;
  const int tr = wave & 3;          // tile row (16 rows)
  const int cb = (wave >> 2) * 4;   // first tile col of this wave (4 tiles)
  v8f acc[4];
#pragma unroll
  for (int t = 0; t < 4; ++t) acc[t] = vzero8();
#pragma unroll
  for (int kk = 0; kk < DIM; kk += 32) {
    const v16bf a = frag_A(sX, tr * 16, kk, DIM, lane);
#pragma unroll
    for (int t = 0; t < 4; ++t) {
      const v16bf b = frag_BT(sW, (cb + t) * 16, kk, DIM, lane);
      acc[t] = wmma_bf16(a, b, acc[t]);
    }
  }
  const float inv_sqrt_d = 0.088388347648318447f;  // 1/sqrt(128)
  const int n = lane & 15, hl = lane >> 4;
#pragma unroll
  for (int t = 0; t < 4; ++t) {
#pragma unroll
    for (int r8 = 0; r8 < 8; ++r8) {
      const int m = tr * 16 + r8 + 8 * hl;
      const float v = __expf(acc[t][r8] - 0.5f * snorm[m]) * inv_sqrt_d;
      phiOut[(rowbase + m) * DIM + (cb + t) * 16 + n] = f2bf(v);
    }
  }
}

// ---------------------------------------------------------------------------
// Kernel: per-chunk state contributions  S_c = phiK^T @ V  (128x128, K=128),
// n_c = column sums of phiK.  One block per (batch, chunk).  Tiles staged
// into LDS by the Tensor Data Mover (issued by wave 0 only).
// ---------------------------------------------------------------------------
__global__ void __launch_bounds__(256) perf_chunksum(
    const unsigned short* __restrict__ phiK,
    const unsigned short* __restrict__ Vb, float* __restrict__ Ssum,
    float* __restrict__ nsum) {
  __shared__ unsigned short sK[CH * DIM];   // 32 KB
  __shared__ unsigned short sV[CH * DIM];   // 32 KB
  const int tid = (int)threadIdx.x;
  const int bc = (int)blockIdx.x;
  const long base = (long)bc * CH * DIM;
  if (tid < 32) {  // wave 0 issues both DMA descriptors, then drains them
    tdm_load_chunk(phiK + base, lds_addr_of(sK));
    tdm_load_chunk(Vb + base, lds_addr_of(sV));
    __builtin_amdgcn_s_wait_tensorcnt(0);
  }
  __syncthreads();
  if (tid < DIM) {  // feature sums over tokens
    float s = 0.f;
    for (int t = 0; t < CH; ++t) s += bf2f(sK[t * DIM + tid]);
    nsum[bc * DIM + tid] = s;
  }
  const int wave = tid >> 5, lane = tid & 31, tr = wave;
  v8f acc[8];
#pragma unroll
  for (int t = 0; t < 8; ++t) acc[t] = vzero8();
#pragma unroll
  for (int kk = 0; kk < CH; kk += 32) {
    const v16bf a = frag_AT(sK, tr * 16, kk, DIM, lane);  // A[m][k]=phiK[k][m]
#pragma unroll
    for (int tc = 0; tc < 8; ++tc) {
      const v16bf b = frag_B(sV, kk, tc * 16, DIM, lane);
      acc[tc] = wmma_bf16(a, b, acc[tc]);
    }
  }
  float* out = Ssum + (long)bc * DIM * DIM;
  const int n = lane & 15, hl = lane >> 4;
#pragma unroll
  for (int tc = 0; tc < 8; ++tc)
#pragma unroll
    for (int r8 = 0; r8 < 8; ++r8)
      out[(tr * 16 + r8 + 8 * hl) * DIM + tc * 16 + n] = acc[tc][r8];
}

// ---------------------------------------------------------------------------
// Kernels: exclusive prefix over chunks (65536 independent 16-long scans for
// M; 512 for N).  Mprev written directly as bf16 for the GEMM consumer.
// ---------------------------------------------------------------------------
__global__ void __launch_bounds__(256) perf_prefixM(
    const float* __restrict__ Ssum, unsigned short* __restrict__ Mprevb) {
  const int idx = blockIdx.x * 256 + (int)threadIdx.x;  // (b, elem)
  const int b = idx >> 14, e = idx & 16383;
  float acc = 0.f;
#pragma unroll
  for (int c = 0; c < NCK; ++c) {
    const long o = (long)(b * NCK + c) * (DIM * DIM) + e;
    Mprevb[o] = f2bf(acc);
    acc += Ssum[o];
  }
}

__global__ void __launch_bounds__(256) perf_prefixN(
    const float* __restrict__ nsum, float* __restrict__ Nprev) {
  const int idx = blockIdx.x * 256 + (int)threadIdx.x;
  if (idx >= NB * DIM) return;
  const int b = idx >> 7, d = idx & 127;
  float acc = 0.f;
#pragma unroll
  for (int c = 0; c < NCK; ++c) {
    const int o = (b * NCK + c) * DIM + d;
    Nprev[o] = acc;
    acc += nsum[o];
  }
}

// ---------------------------------------------------------------------------
// Kernel: per-chunk output.
//   scores = causal_mask(phiQ @ phiK^T)            (GEMM1, 128x128,K=128)
//   out    = scores @ V + phiQ @ Mprev             (GEMM2a + GEMM2b)
//   r_i    = dot(phiQ_i, Nprev) + rowsum(scores_i) ; out /= r + sign(r)*1e-6
// One block per (batch, chunk); 64 KB LDS ping-ponged across phases with TDM
// staging (V-tile DMA overlaps the masked-score epilogue).
// ---------------------------------------------------------------------------
__global__ void __launch_bounds__(256) perf_output(
    const unsigned short* __restrict__ phiQ,
    const unsigned short* __restrict__ phiK,
    const unsigned short* __restrict__ Vb,
    const unsigned short* __restrict__ Mprevb,
    const float* __restrict__ Nprev, float* __restrict__ rbuf,
    float* __restrict__ Out) {
  __shared__ unsigned short bufA[CH * DIM];  // 32 KB
  __shared__ unsigned short bufB[CH * DIM];  // 32 KB
  const int tid = (int)threadIdx.x, wave = tid >> 5, lane = tid & 31;
  const int tr = wave;
  const int bc = (int)blockIdx.x;
  const long base = (long)bc * CH * DIM;

  // phase 0: TDM  phiQ -> bufA, phiK -> bufB
  if (tid < 32) {
    tdm_load_chunk(phiQ + base, lds_addr_of(bufA));
    tdm_load_chunk(phiK + base, lds_addr_of(bufB));
    __builtin_amdgcn_s_wait_tensorcnt(0);
  }
  __syncthreads();

  // inter-chunk denominator part: dot(phiQ_i, Nprev)
  if (tid < CH) {
    float s = 0.f;
    const float* np = Nprev + bc * DIM;
    for (int d = 0; d < DIM; ++d) s += bf2f(bufA[tid * DIM + d]) * np[d];
    rbuf[bc * CH + tid] = s;
  }

  // GEMM1: scores[i][j] = sum_d phiQ[i][d]*phiK[j][d]; skip future tiles
  v8f acc1[8];
#pragma unroll
  for (int t = 0; t < 8; ++t) acc1[t] = vzero8();
#pragma unroll
  for (int kk = 0; kk < DIM; kk += 32) {
    const v16bf a = frag_A(bufA, tr * 16, kk, DIM, lane);
#pragma unroll
    for (int tc = 0; tc < 8; ++tc) {
      if (tc > tr) continue;  // causal: strictly-future tiles are zero
      const v16bf b = frag_BT(bufB, tc * 16, kk, DIM, lane);
      acc1[tc] = wmma_bf16(a, b, acc1[tc]);
    }
  }
  __syncthreads();  // all reads of bufA/bufB done

  // kick off async V -> bufA NOW; it overlaps the masked-score epilogue
  if (tid < 32) tdm_load_chunk(Vb + base, lds_addr_of(bufA));

  // write causally-masked scores (bf16) into bufB; prefetch GEMM2b operands
  const int n = lane & 15, hl = lane >> 4;
#pragma unroll
  for (int tc = 0; tc < 8; ++tc) {
#pragma unroll
    for (int r8 = 0; r8 < 8; ++r8) {
      const int i = tr * 16 + r8 + 8 * hl;
      const int j = tc * 16 + n;
      const float v =
          (tc < tr) ? acc1[tc][r8]
                    : ((tc == tr && j <= i) ? acc1[tc][r8] : 0.f);
      bufB[i * DIM + j] = f2bf(v);
    }
  }
  __builtin_prefetch(&phiQ[base + tid * 64], 0, 1);    // global_prefetch_b8
  __builtin_prefetch(&Mprevb[base + tid * 64], 0, 1);
  __syncthreads();

  // intra-chunk denominator part: row sums of masked scores
  if (tid < CH) {
    float s = 0.f;
    for (int j = 0; j < CH; ++j) s += bf2f(bufB[tid * DIM + j]);
    rbuf[bc * CH + tid] += s;
  }
  if (tid < 32) __builtin_amdgcn_s_wait_tensorcnt(0);  // V tile landed
  __syncthreads();

  // GEMM2a: out += scores @ V ; K blocks beyond the diagonal tile are zero
  v8f acc2[8];
#pragma unroll
  for (int t = 0; t < 8; ++t) acc2[t] = vzero8();
#pragma unroll
  for (int kk = 0; kk < CH; kk += 32) {
    if (kk > tr * 16 + 15) continue;  // masked region is all-zero
    const v16bf a = frag_A(bufB, tr * 16, kk, DIM, lane);
#pragma unroll
    for (int tc = 0; tc < 8; ++tc) {
      const v16bf b = frag_B(bufA, kk, tc * 16, DIM, lane);
      acc2[tc] = wmma_bf16(a, b, acc2[tc]);
    }
  }
  __syncthreads();  // GEMM2a reads done

  // TDM  phiQ -> bufA (reload), Mprev -> bufB
  if (tid < 32) {
    tdm_load_chunk(phiQ + base, lds_addr_of(bufA));
    tdm_load_chunk(Mprevb + base, lds_addr_of(bufB));
    __builtin_amdgcn_s_wait_tensorcnt(0);
  }
  __syncthreads();

  // GEMM2b: out += phiQ @ Mprev
#pragma unroll
  for (int kk = 0; kk < DIM; kk += 32) {
    const v16bf a = frag_A(bufA, tr * 16, kk, DIM, lane);
#pragma unroll
    for (int tc = 0; tc < 8; ++tc) {
      const v16bf b = frag_B(bufB, kk, tc * 16, DIM, lane);
      acc2[tc] = wmma_bf16(a, b, acc2[tc]);
    }
  }

  // epilogue: divide by denominator, store f32
#pragma unroll
  for (int tc = 0; tc < 8; ++tc) {
#pragma unroll
    for (int r8 = 0; r8 < 8; ++r8) {
      const int i = tr * 16 + r8 + 8 * hl;
      const float r = rbuf[bc * CH + i];
      const float den = r + copysignf(1e-6f, r);
      Out[base + i * DIM + tc * 16 + n] = acc2[tc][r8] / den;
    }
  }
}

// ---------------------------------------------------------------------------
// Host side
// ---------------------------------------------------------------------------
extern "C" void kernel_launch(void* const* d_in, const int* in_sizes, int n_in,
                              void* d_out, int out_size, void* d_ws,
                              size_t ws_size, hipStream_t stream) {
  const float* K = (const float*)d_in[0];
  const float* Q = (const float*)d_in[1];
  const float* V = (const float*)d_in[2];
  const float* W = (const float*)d_in[6];
  float* Out = (float*)d_out;
  char* ws = (char*)d_ws;

  // workspace layout (bytes)
  unsigned short* phiKb  = (unsigned short*)(ws);              //  2 MB
  unsigned short* phiQb  = (unsigned short*)(ws + 2097152);    //  2 MB
  unsigned short* Vb     = (unsigned short*)(ws + 4194304);    //  2 MB
  float*          Ssum   = (float*)(ws + 6291456);             //  4 MB
  float*          nsum   = (float*)(ws + 10485760);            // 32 KB
  unsigned short* Mprevb = (unsigned short*)(ws + 10518528);   //  2 MB
  float*          Nprev  = (float*)(ws + 12615680);            // 32 KB
  float*          rbuf   = (float*)(ws + 12648448);            // 32 KB

  const int total = NB * NS * DIM;  // 1,048,576 elements

  perf_cast_bf16<<<(total + 255) / 256, 256, 0, stream>>>(V, Vb, total);
  perf_featurize<<<NB * NS / 64, 256, 0, stream>>>(K, W, phiKb);
  perf_featurize<<<NB * NS / 64, 256, 0, stream>>>(Q, W, phiQb);
  perf_chunksum<<<NB * NCK, 256, 0, stream>>>(phiKb, Vb, Ssum, nsum);
  perf_prefixM<<<NB * DIM * DIM / 256, 256, 0, stream>>>(Ssum, Mprevb);
  perf_prefixN<<<2, 256, 0, stream>>>(nsum, Nprev);
  perf_output<<<NB * NCK, 256, 0, stream>>>(phiQb, phiKb, Vb, Mprevb, Nprev,
                                            rbuf, Out);
}
